// ProposalDistribution_53790170415485
// MI455X (gfx1250) — matrix operand order repaired and checked by reading, not compiled, not run on hardware
//
#include <hip/hip_runtime.h>
#include <stdint.h>

typedef __attribute__((ext_vector_type(16))) __bf16 v16bf;
typedef __attribute__((ext_vector_type(8)))  float  v8f;

#define K_DIM   512
#define H_DIM   512
#define M_TILE  64
#define THREADS 512
#define ASTRIDE 520          // padded ushort row stride for LDS A tile (bank-conflict free)
#define LOG_EPS (-13.815511f)
#define EPSV    1e-6f
#define WMAT    ((size_t)K_DIM * H_DIM)   // elements per weight matrix

union BF16x16 {
    v16bf v;
    unsigned short u[16];
    unsigned int   w[8];
    uint4 q[2];
};

// native packed f32 -> bf16 (compiler should select v_cvt_pk_bf16_f32-class ops)
static __device__ inline unsigned int pk_bf16(float a, float b) {
    union { __bf16 h[2]; unsigned int u; } r;
    r.h[0] = (__bf16)a;
    r.h[1] = (__bf16)b;
    return r.u;
}

static __device__ inline float rng01(unsigned int& st) {
    st = st * 1664525u + 1013904223u;
    return (float)(st >> 8) * (1.0f / 16777216.0f);
}

// One-shot weight conversion: 4 x [512,512] f32 -> bf16 in workspace.
__global__ __launch_bounds__(256)
void convert_weights_kernel(const float* __restrict__ w0, const float* __restrict__ w1,
                            const float* __restrict__ w2, const float* __restrict__ w3,
                            unsigned short* __restrict__ dst)
{
    size_t i = (size_t)blockIdx.x * blockDim.x + threadIdx.x;   // one float4 per thread
    size_t mat = i >> 16;                                       // 65536 float4 per matrix
    size_t off = (i & 65535u) * 4;
    const float* s = (mat == 0) ? w0 : (mat == 1) ? w1 : (mat == 2) ? w2 : w3;
    float4 f = *(const float4*)(s + off);
    uint2 o;
    o.x = pk_bf16(f.x, f.y);
    o.y = pk_bf16(f.z, f.w);
    *(uint2*)(dst + mat * WMAT + off) = o;
}

extern __shared__ unsigned char smem_raw[];

template <bool PRECONV>
__global__ __launch_bounds__(THREADS, 1)
void proposal_fused_kernel(const float* __restrict__ x,    const float* __restrict__ y,
                           const float* __restrict__ Wmux, const float* __restrict__ bmux,
                           const float* __restrict__ Wsgx, const float* __restrict__ bsgx,
                           const float* __restrict__ Wmuy, const float* __restrict__ bmuy,
                           const float* __restrict__ Wsgy, const float* __restrict__ bsgy,
                           const unsigned short* __restrict__ Wb,   // bf16 weights (if PRECONV)
                           float* __restrict__ out)
{
    unsigned short* sA   = (unsigned short*)smem_raw;                  // 64 x 520 bf16
    float*          sums = (float*)(smem_raw + M_TILE * ASTRIDE * 2);  // 64 rows x 8 sums

    const int tid  = threadIdx.x;
    const int wave = tid >> 5;
    const int lane = tid & 31;
    const int lh   = lane >> 4;        // lane half (0/1)
    const int ln   = lane & 15;
    const int rowBlock = blockIdx.x * M_TILE;
    const int nB = wave * 32;          // this wave's 32 output columns

    sums[tid] = 0.0f;                  // 512 floats, one per thread

    v8f c_mu[4][2];
    v8f c_sg[4][2];

    for (int phase = 0; phase < 2; ++phase) {
        const float* A   = phase ? y    : x;
        const float* Wmu = phase ? Wmuy : Wmux;
        const float* bmu = phase ? bmuy : bmux;
        const float* Wsg = phase ? Wsgy : Wsgx;
        const float* bsg = phase ? bsgy : bsgx;
        const unsigned short* WbMu = Wb + (size_t)(phase * 2 + 0) * WMAT;
        const unsigned short* WbSg = Wb + (size_t)(phase * 2 + 1) * WMAT;

        __syncthreads();
        // Stage A tile: 64 rows x 512 f32 -> bf16 in LDS (coalesced float4 loads)
        for (int it = tid; it < (M_TILE * K_DIM / 4); it += THREADS) {
            int idx = it * 4;
            int m = idx >> 9;
            int k = idx & 511;
            float4 f = *(const float4*)(A + (size_t)(rowBlock + m) * K_DIM + k);
            uint2 st;
            st.x = pk_bf16(f.x, f.y);
            st.y = pk_bf16(f.z, f.w);
            *(uint2*)(sA + m * ASTRIDE + k) = st;
        }
        __syncthreads();

        #pragma unroll
        for (int mt = 0; mt < 4; ++mt)
            #pragma unroll
            for (int nt = 0; nt < 2; ++nt)
                #pragma unroll
                for (int e = 0; e < 8; ++e) { c_mu[mt][nt][e] = 0.0f; c_sg[mt][nt][e] = 0.0f; }

        for (int kb = 0; kb < K_DIM; kb += 32) {
            // A fragments for the 4 M-tiles (contiguous K = kb+lh*16..+15, matching B)
            BF16x16 a[4];
            const int abase = kb + lh * 16;
            #pragma unroll
            for (int mt = 0; mt < 4; ++mt) {
                const unsigned short* p = sA + (mt * 16 + ln) * ASTRIDE + abase;
                a[mt].q[0] = *(const uint4*)(p);
                a[mt].q[1] = *(const uint4*)(p + 8);
            }
            #pragma unroll
            for (int nt = 0; nt < 2; ++nt) {
                const int n = nB + nt * 16 + ln;
                const size_t woff = (size_t)n * K_DIM + kb + lh * 16;  // 16 contiguous K elems
                BF16x16 bmu16, bsg16;
                if (PRECONV) {
                    const uint4* pm = (const uint4*)(WbMu + woff);
                    const uint4* ps = (const uint4*)(WbSg + woff);
                    uint4 m0 = pm[0], m1 = pm[1];      // issue all loads up front
                    uint4 s0 = ps[0], s1 = ps[1];
                    bmu16.q[0] = m0; bmu16.q[1] = m1;
                    bsg16.q[0] = s0; bsg16.q[1] = s1;
                } else {
                    const float4* pm = (const float4*)(Wmu + woff);
                    const float4* ps = (const float4*)(Wsg + woff);
                    float4 m0 = pm[0], m1 = pm[1], m2 = pm[2], m3 = pm[3];
                    float4 s0 = ps[0], s1 = ps[1], s2 = ps[2], s3 = ps[3];
                    bmu16.w[0] = pk_bf16(m0.x, m0.y); bmu16.w[1] = pk_bf16(m0.z, m0.w);
                    bmu16.w[2] = pk_bf16(m1.x, m1.y); bmu16.w[3] = pk_bf16(m1.z, m1.w);
                    bmu16.w[4] = pk_bf16(m2.x, m2.y); bmu16.w[5] = pk_bf16(m2.z, m2.w);
                    bmu16.w[6] = pk_bf16(m3.x, m3.y); bmu16.w[7] = pk_bf16(m3.z, m3.w);
                    bsg16.w[0] = pk_bf16(s0.x, s0.y); bsg16.w[1] = pk_bf16(s0.z, s0.w);
                    bsg16.w[2] = pk_bf16(s1.x, s1.y); bsg16.w[3] = pk_bf16(s1.z, s1.w);
                    bsg16.w[4] = pk_bf16(s2.x, s2.y); bsg16.w[5] = pk_bf16(s2.z, s2.w);
                    bsg16.w[6] = pk_bf16(s3.x, s3.y); bsg16.w[7] = pk_bf16(s3.z, s3.w);
                }
                #pragma unroll
                for (int mt = 0; mt < 4; ++mt)
                    c_mu[mt][nt] = __builtin_amdgcn_wmma_f32_16x16x32_bf16(
                        false, a[mt].v, false, bmu16.v, (short)0, c_mu[mt][nt], false, false);
                #pragma unroll
                for (int mt = 0; mt < 4; ++mt)
                    c_sg[mt][nt] = __builtin_amdgcn_wmma_f32_16x16x32_bf16(
                        false, a[mt].v, false, bsg16.v, (short)0, c_sg[mt][nt], false, false);
            }
        }

        // Epilogue: bias, exp, and the four per-row reductions
        #pragma unroll
        for (int mt = 0; mt < 4; ++mt) {
            #pragma unroll
            for (int r = 0; r < 8; ++r) {
                float s0 = 0.f, s1 = 0.f, s2 = 0.f, s3 = 0.f;
                #pragma unroll
                for (int nt = 0; nt < 2; ++nt) {
                    const int n = nB + nt * 16 + ln;
                    float mean = c_mu[mt][nt][r] + bmu[n];
                    float z    = c_sg[mt][nt][r] + bsg[n];
                    float v    = __expf(z);
                    float inv  = 1.0f / fmaxf(v, EPSV);
                    s0 += fmaxf(z, LOG_EPS);      // log(max(exp(z), eps))
                    s1 += inv;
                    s2 += mean * inv;
                    s3 += mean * mean * inv;
                }
                #pragma unroll
                for (int msk = 8; msk >= 1; msk >>= 1) {
                    s0 += __shfl_xor(s0, msk, 32);
                    s1 += __shfl_xor(s1, msk, 32);
                    s2 += __shfl_xor(s2, msk, 32);
                    s3 += __shfl_xor(s3, msk, 32);
                }
                if (ln == 0) {
                    int row = mt * 16 + lh * 8 + r;
                    float* d = sums + row * 8 + phase * 4;
                    atomicAdd(d + 0, s0);
                    atomicAdd(d + 1, s1);
                    atomicAdd(d + 2, s2);
                    atomicAdd(d + 3, s3);
                }
            }
        }
    }

    __syncthreads();
    // Rejection sampling: one thread per row; q expanded from the 8 per-row sums.
    if (tid < M_TILE) {
        const int row = rowBlock + tid;
        const float* s = sums + tid * 8;
        float s0x = s[0], s1x = s[1], s2x = s[2], s3x = s[3];
        float s0y = s[4], s1y = s[5], s2y = s[6], s3y = s[7];
        const float sc = 0.5f / (float)H_DIM;
        unsigned int st = (unsigned int)row * 0x9E3779B9u + 12345u;
        float xi = 0.0f;
        for (int it = 0; it < 64; ++it) {
            float u = rng01(st);
            xi = 2.0f * rng01(st) - 1.0f;
            float qx = sc * (s0x + xi * xi * s1x - 2.0f * xi * s2x + s3x);
            float qy = sc * (s0y + xi * xi * s1y - 2.0f * xi * s2y + s3y);
            if (u >= qx * qy) break;
        }
        out[row] = fminf(fmaxf(xi, 1e-5f), 10.0f);
    }
}

extern "C" void kernel_launch(void* const* d_in, const int* in_sizes, int n_in,
                              void* d_out, int out_size, void* d_ws, size_t ws_size,
                              hipStream_t stream) {
    (void)n_in; (void)out_size;
    const float* x    = (const float*)d_in[0];
    const float* y    = (const float*)d_in[1];
    const float* Wmux = (const float*)d_in[2];
    const float* bmux = (const float*)d_in[3];
    const float* Wsgx = (const float*)d_in[4];
    const float* bsgx = (const float*)d_in[5];
    const float* Wmuy = (const float*)d_in[6];
    const float* bmuy = (const float*)d_in[7];
    const float* Wsgy = (const float*)d_in[8];
    const float* bsgy = (const float*)d_in[9];
    float* out = (float*)d_out;

    const int rows = in_sizes[0] / K_DIM;      // 32768
    const int grid = rows / M_TILE;            // 512 blocks
    const size_t shmem = (size_t)M_TILE * ASTRIDE * 2 + (size_t)THREADS * 4;  // 68608 B
    const size_t wbytes = 4 * WMAT * sizeof(unsigned short);                   // 2 MB

    if (ws_size >= wbytes) {
        unsigned short* Wb = (unsigned short*)d_ws;
        // bf16-ify all four weight matrices into L2-resident scratch (order: mux, sgx, muy, sgy)
        convert_weights_kernel<<<(4 * WMAT / 4) / 256, 256, 0, stream>>>(
            Wmux, Wsgx, Wmuy, Wsgy, Wb);
        hipFuncSetAttribute(reinterpret_cast<const void*>(proposal_fused_kernel<true>),
                            hipFuncAttributeMaxDynamicSharedMemorySize, (int)shmem);
        proposal_fused_kernel<true><<<grid, THREADS, shmem, stream>>>(
            x, y, Wmux, bmux, Wsgx, bsgx, Wmuy, bmuy, Wsgy, bsgy, Wb, out);
    } else {
        hipFuncSetAttribute(reinterpret_cast<const void*>(proposal_fused_kernel<false>),
                            hipFuncAttributeMaxDynamicSharedMemorySize, (int)shmem);
        proposal_fused_kernel<false><<<grid, THREADS, shmem, stream>>>(
            x, y, Wmux, bmux, Wsgx, bsgx, Wmuy, bmuy, Wsgy, bsgy, nullptr, out);
    }
}
